// SelfAttentionLayer_8684423873165
// MI455X (gfx1250) — matrix operand order repaired and checked by reading, not compiled, not run on hardware
//
#include <hip/hip_runtime.h>

// ---------------------------------------------------------------------------
// MI455X / gfx1250 self-attention (B=2, S=4096, H=2048, fp32 ref).
// Compute-bound (~0.48 TFLOP vs ~15us HBM) -> all five matmuls through
// v_wmma_f32_16x16x32_bf16. bf16-operand GEMMs (scores, P@V) stage tiles with
// double-buffered GLOBAL_LOAD_ASYNC_TO_LDS_B128 (ASYNCcnt); f32-operand GEMMs
// (QKV projections) use load+cvt_pk_bf16+ds_store staging.
// ---------------------------------------------------------------------------

typedef float  v8f   __attribute__((ext_vector_type(8)));
typedef __bf16 v16bf __attribute__((ext_vector_type(16)));
typedef __bf16 v8bf  __attribute__((ext_vector_type(8)));
typedef int    v4i   __attribute__((ext_vector_type(4)));

typedef __attribute__((address_space(1))) v4i* as1_v4i;
typedef __attribute__((address_space(3))) v4i* as3_v4i;

#define TILE_M 128
#define TILE_N 128
#define TILE_K 32
#define LDS_STRIDE 40   // halves per row: 32 data + 8 pad, keeps 16B alignment

#if __has_builtin(__builtin_amdgcn_global_load_async_to_lds_b128)
#define ASYNC_LDS 1
#else
#define ASYNC_LDS 0
#endif

#if __has_builtin(__builtin_amdgcn_s_wait_asynccnt)
#define WAIT_ASYNC(n) __builtin_amdgcn_s_wait_asynccnt(n)
#else
#define WAIT_ASYNC(n) asm volatile("s_wait_asynccnt %0" ::"n"(n))
#endif

union BfFrag { v16bf v; v8bf h[2]; };

// Generic tiled GEMM: C[m,n] = sum_k A[m,k] * B[n,k]  (+ bias[n])
//   A: M x K rows (f32 or bf16), B: N x K rows (f32 or bf16)
//   CMODE: 0 = f32 C[m*ldc+n], 1 = bf16 C[m*ldc+n], 2 = bf16 C[n*ldc+m]
// All strides are in BYTES. M,N multiples of 128; K multiple of 32.
template<bool AF32, bool BF32, int CMODE, bool HAS_BIAS>
__launch_bounds__(256)
__global__ void gemm_bt_wmma(const char* __restrict__ A, size_t ldaB, size_t bsA,
                             const char* __restrict__ Bm, size_t ldbB, size_t bsB,
                             const float* __restrict__ bias,
                             char* __restrict__ C, size_t ldcB, size_t bsC,
                             int K)
{
    constexpr bool USE_ASYNC = (!AF32) && (!BF32) && (ASYNC_LDS != 0);
    constexpr int  NBUF      = USE_ASYNC ? 2 : 1;
    constexpr int  BUF_HALVES = TILE_M * LDS_STRIDE;

    __shared__ __bf16 As[NBUF * BUF_HALVES];
    __shared__ __bf16 Bs[NBUF * BUF_HALVES];

    const int t    = threadIdx.x;
    const int lane = t & 31;           // wave32
    const int wave = t >> 5;           // 8 waves
    const int wm   = wave & 1;         // 2 waves over M
    const int wn   = wave >> 1;        // 4 waves over N
    const int mBase = wm * 64;
    const int nBase = wn * 32;
    const int llo  = lane & 15;
    const int lhi  = lane >> 4;

    const char* Ab = A  + (size_t)blockIdx.z * bsA;
    const char* Bb = Bm + (size_t)blockIdx.z * bsB;
    char*       Cb = C  + (size_t)blockIdx.z * bsC;

    const int m0 = blockIdx.y * TILE_M;
    const int n0 = blockIdx.x * TILE_N;

    // cooperative staging: 256 threads, each stages one 16-element half-row
    const int lrow  = t >> 1;          // 0..127
    const int lcolh = (t & 1) * 16;    // 0 or 16

    v8f acc[4][2];
    #pragma unroll
    for (int mt = 0; mt < 4; ++mt)
        #pragma unroll
        for (int nt = 0; nt < 2; ++nt)
            #pragma unroll
            for (int i = 0; i < 8; ++i)
                acc[mt][nt][i] = 0.0f;

    // ---- fragment gather + 8 WMMAs from LDS buffer at half-offset bufOff ----
    auto compute_tile = [&](int bufOff) {
        // A 16x32: lanes 0-15 -> M=lane, K={0..7,16..23}; lanes 16-31 -> K={8..15,24..31}
        BfFrag afrag[4];
        #pragma unroll
        for (int mt = 0; mt < 4; ++mt) {
            const __bf16* ap = &As[bufOff + (mBase + mt*16 + llo) * LDS_STRIDE + lhi * 8];
            afrag[mt].h[0] = *(const v8bf*)ap;
            afrag[mt].h[1] = *(const v8bf*)(ap + 16);
        }
        // B 32x16: lanes 0-15 -> N=lane, K={0..7,16..23}; lanes 16-31 -> K={8..15,24..31}
        BfFrag bfrag[2];
        #pragma unroll
        for (int nt = 0; nt < 2; ++nt) {
            const __bf16* bp = &Bs[bufOff + (nBase + nt*16 + llo) * LDS_STRIDE + lhi * 8];
            bfrag[nt].h[0] = *(const v8bf*)bp;
            bfrag[nt].h[1] = *(const v8bf*)(bp + 16);
        }
        #pragma unroll
        for (int mt = 0; mt < 4; ++mt)
            #pragma unroll
            for (int nt = 0; nt < 2; ++nt)
                acc[mt][nt] = __builtin_amdgcn_wmma_f32_16x16x32_bf16(
                    false, afrag[mt].v, false, bfrag[nt].v,
                    (short)0, acc[mt][nt], false, false);
    };

    if constexpr (USE_ASYNC) {
        // bf16 operands: double-buffered async copy global->LDS (ASYNCcnt).
        // Per stage each thread issues 4 async b128 ops (2 for A, 2 for B).
        auto stage = [&](int k0, int buf) {
            const char* pa = Ab + (size_t)(m0 + lrow) * ldaB + (size_t)(k0 + lcolh) * 2;
            const char* pb = Bb + (size_t)(n0 + lrow) * ldbB + (size_t)(k0 + lcolh) * 2;
            __bf16* da = &As[buf * BUF_HALVES + lrow * LDS_STRIDE + lcolh];
            __bf16* db = &Bs[buf * BUF_HALVES + lrow * LDS_STRIDE + lcolh];
            // INST_OFFSET is added to BOTH the global and the LDS address.
            __builtin_amdgcn_global_load_async_to_lds_b128((as1_v4i)pa, (as3_v4i)da, 0,  0);
            __builtin_amdgcn_global_load_async_to_lds_b128((as1_v4i)pa, (as3_v4i)da, 16, 0);
            __builtin_amdgcn_global_load_async_to_lds_b128((as1_v4i)pb, (as3_v4i)db, 0,  0);
            __builtin_amdgcn_global_load_async_to_lds_b128((as1_v4i)pb, (as3_v4i)db, 16, 0);
        };
        stage(0, 0);
        int cur = 0;
        for (int k0 = 0; k0 < K; k0 += TILE_K) {
            const bool hasNext = (k0 + TILE_K) < K;
            if (hasNext) {
                stage(k0 + TILE_K, cur ^ 1);   // overlap next tile with compute
                WAIT_ASYNC(4);                 // own stage(cur) complete; next in flight
            } else {
                WAIT_ASYNC(0);
            }
            __syncthreads();                   // all waves' stage(cur) complete
            compute_tile(cur * BUF_HALVES);
            __syncthreads();                   // cur buffer free for reuse
            cur ^= 1;
        }
    } else {
        // f32 (or fallback) operands: load -> cvt_pk_bf16 -> ds_store staging.
        for (int k0 = 0; k0 < K; k0 += TILE_K) {
            {
                const size_t esz = AF32 ? 4u : 2u;
                const char* p = Ab + (size_t)(m0 + lrow) * ldaB + (size_t)(k0 + lcolh) * esz;
                __bf16* dst = &As[lrow * LDS_STRIDE + lcolh];
                if (AF32) {
                    const float4* q = (const float4*)p;
                    float4 x0 = q[0], x1 = q[1], x2 = q[2], x3 = q[3];
                    float xs[16] = { x0.x,x0.y,x0.z,x0.w, x1.x,x1.y,x1.z,x1.w,
                                     x2.x,x2.y,x2.z,x2.w, x3.x,x3.y,x3.z,x3.w };
                    v8bf h0, h1;
                    #pragma unroll
                    for (int i = 0; i < 8; ++i) { h0[i] = (__bf16)xs[i]; h1[i] = (__bf16)xs[8+i]; }
                    *(v8bf*)dst = h0; *(v8bf*)(dst + 8) = h1;
                } else {
                    const v8bf* q = (const v8bf*)p;
                    v8bf h0 = q[0], h1 = q[1];
                    *(v8bf*)dst = h0; *(v8bf*)(dst + 8) = h1;
                }
                if (k0 + TILE_K < K) __builtin_prefetch(p + TILE_K * esz, 0, 3);
            }
            {
                const size_t esz = BF32 ? 4u : 2u;
                const char* p = Bb + (size_t)(n0 + lrow) * ldbB + (size_t)(k0 + lcolh) * esz;
                __bf16* dst = &Bs[lrow * LDS_STRIDE + lcolh];
                if (BF32) {
                    const float4* q = (const float4*)p;
                    float4 x0 = q[0], x1 = q[1], x2 = q[2], x3 = q[3];
                    float xs[16] = { x0.x,x0.y,x0.z,x0.w, x1.x,x1.y,x1.z,x1.w,
                                     x2.x,x2.y,x2.z,x2.w, x3.x,x3.y,x3.z,x3.w };
                    v8bf h0, h1;
                    #pragma unroll
                    for (int i = 0; i < 8; ++i) { h0[i] = (__bf16)xs[i]; h1[i] = (__bf16)xs[8+i]; }
                    *(v8bf*)dst = h0; *(v8bf*)(dst + 8) = h1;
                } else {
                    const v8bf* q = (const v8bf*)p;
                    v8bf h0 = q[0], h1 = q[1];
                    *(v8bf*)dst = h0; *(v8bf*)(dst + 8) = h1;
                }
                if (k0 + TILE_K < K) __builtin_prefetch(p + TILE_K * esz, 0, 3);
            }
            __syncthreads();
            compute_tile(0);
            __syncthreads();
        }
    }

    // ---- epilogue: C/D layout: VGPR r, lanes 0-15 -> M=r, lanes 16-31 -> M=8+r; N=lane%16
    #pragma unroll
    for (int mt = 0; mt < 4; ++mt) {
        #pragma unroll
        for (int nt = 0; nt < 2; ++nt) {
            const int gm0 = m0 + mBase + mt*16 + lhi*8;
            const int gn  = n0 + nBase + nt*16 + llo;
            float bval = 0.0f;
            if (HAS_BIAS) bval = bias[gn];
            #pragma unroll
            for (int r = 0; r < 8; ++r) {
                const int gm = gm0 + r;
                const float val = acc[mt][nt][r] + bval;
                if (CMODE == 0) {
                    *(float*)(Cb + (size_t)gm * ldcB + (size_t)gn * 4) = val;
                } else if (CMODE == 1) {
                    *(__bf16*)(Cb + (size_t)gm * ldcB + (size_t)gn * 2) = (__bf16)val;
                } else {
                    *(__bf16*)(Cb + (size_t)gn * ldcB + (size_t)gm * 2) = (__bf16)val;
                }
            }
        }
    }
}

// In-place row softmax over a bf16 [rows x rowLen] matrix. One block per row,
// 256 threads, rowLen = 4096 -> 16 elements per thread. Each block reads its
// full row before writing, so in-place is safe.
__launch_bounds__(256)
__global__ void softmax_rows_bf16(__bf16* __restrict__ P, int rowLen)
{
    const int t = threadIdx.x;
    __bf16* row = P + (size_t)blockIdx.x * rowLen;
    const int base = t * 16;

    v8bf a = *(const v8bf*)(row + base);
    v8bf b = *(const v8bf*)(row + base + 8);
    float x[16];
    #pragma unroll
    for (int i = 0; i < 8; ++i) { x[i] = (float)a[i]; x[8+i] = (float)b[i]; }

    float mx = x[0];
    #pragma unroll
    for (int i = 1; i < 16; ++i) mx = fmaxf(mx, x[i]);

    __shared__ float red[256];
    red[t] = mx; __syncthreads();
    for (int s2 = 128; s2 > 0; s2 >>= 1) {
        if (t < s2) red[t] = fmaxf(red[t], red[t + s2]);
        __syncthreads();
    }
    mx = red[0]; __syncthreads();

    float sum = 0.0f;
    #pragma unroll
    for (int i = 0; i < 16; ++i) { x[i] = __expf(x[i] - mx); sum += x[i]; }
    red[t] = sum; __syncthreads();
    for (int s2 = 128; s2 > 0; s2 >>= 1) {
        if (t < s2) red[t] = red[t] + red[t + s2];
        __syncthreads();
    }
    const float inv = 1.0f / red[0];

    #pragma unroll
    for (int i = 0; i < 8; ++i) { a[i] = (__bf16)(x[i] * inv); b[i] = (__bf16)(x[8+i] * inv); }
    *(v8bf*)(row + base) = a;
    *(v8bf*)(row + base + 8) = b;
}

extern "C" void kernel_launch(void* const* d_in, const int* in_sizes, int n_in,
                              void* d_out, int out_size, void* d_ws, size_t ws_size,
                              hipStream_t stream)
{
    (void)in_sizes; (void)n_in; (void)out_size; (void)ws_size;
    constexpr int BATCH = 2, S = 4096, H = 2048;

    const char*  X  = (const char*)d_in[0];
    const char*  Wq = (const char*)d_in[1];
    const float* bq = (const float*)d_in[2];
    const char*  Wk = (const char*)d_in[3];
    const float* bk = (const float*)d_in[4];
    const char*  Wv = (const char*)d_in[5];
    const float* bv = (const float*)d_in[6];

    // Workspace layout (bf16): Q[B,S,H], K[B,S,H], Vt[B,H,S], P[B,S,S]  (~168 MB)
    char* ws  = (char*)d_ws;
    const size_t szQKV = (size_t)BATCH * S * H * 2;
    char* Qbf = ws;
    char* Kbf = Qbf + szQKV;
    char* Vt  = Kbf + szQKV;
    char* Pbf = Vt  + szQKV;

    const dim3 blk(256);

    // --- QKV projections: y = x @ W^T + b ---
    const dim3 gProj(H / TILE_N, S / TILE_M, BATCH);
    gemm_bt_wmma<true, true, 1, true><<<gProj, blk, 0, stream>>>(
        X, (size_t)H * 4, (size_t)S * H * 4,
        Wq, (size_t)H * 4, (size_t)0,
        bq,
        Qbf, (size_t)H * 2, (size_t)S * H * 2, H);

    gemm_bt_wmma<true, true, 1, true><<<gProj, blk, 0, stream>>>(
        X, (size_t)H * 4, (size_t)S * H * 4,
        Wk, (size_t)H * 4, (size_t)0,
        bk,
        Kbf, (size_t)H * 2, (size_t)S * H * 2, H);

    // V written transposed: Vt[b][h][s], so P@V sees V as N-major rows of K
    gemm_bt_wmma<true, true, 2, true><<<gProj, blk, 0, stream>>>(
        X, (size_t)H * 4, (size_t)S * H * 4,
        Wv, (size_t)H * 4, (size_t)0,
        bv,
        Vt, (size_t)S * 2, (size_t)H * S * 2, H);

    // --- scores = Q @ K^T (no scaling, per reference), bf16 out ---
    const dim3 gS(S / TILE_N, S / TILE_M, BATCH);
    gemm_bt_wmma<false, false, 1, false><<<gS, blk, 0, stream>>>(
        Qbf, (size_t)H * 2, (size_t)S * H * 2,
        Kbf, (size_t)H * 2, (size_t)S * H * 2,
        nullptr,
        Pbf, (size_t)S * 2, (size_t)S * S * 2, H);

    // --- row softmax, in-place on the bf16 score matrix ---
    softmax_rows_bf16<<<dim3(BATCH * S), blk, 0, stream>>>((__bf16*)Pbf, S);

    // --- out = P @ V : A = P (M=S, K=S), B = Vt (N=H rows of K=S), f32 out ---
    const dim3 gO(H / TILE_N, S / TILE_M, BATCH);
    gemm_bt_wmma<false, false, 0, false><<<gO, blk, 0, stream>>>(
        Pbf, (size_t)S * 2, (size_t)S * S * 2,
        Vt,  (size_t)S * 2, (size_t)H * S * 2,
        nullptr,
        (char*)d_out, (size_t)H * 4, (size_t)S * H * 4, S);
}
